// Model_56375740727940
// MI455X (gfx1250) — compile-verified, matrix operand order checked by reading, not run on hardware
//
#include <hip/hip_runtime.h>
#include <hip/hip_bf16.h>

#define N_NODES 100000
#define N_EDGES 1600000
#define DIM     128
#define NLAYERS 2
#define SCAN_B  256
#define NBLK    ((N_NODES + SCAN_B - 1) / SCAN_B)   // 391
#define ROWSTR  72                                  // LDS row stride in dwords (64 + 8 pad)

typedef __attribute__((ext_vector_type(16))) __bf16       v16bf;
typedef __attribute__((ext_vector_type(8)))  float        v8f;
typedef __attribute__((ext_vector_type(8)))  unsigned int v8u;

__device__ __forceinline__ float warp_sum32(float v) {
#pragma unroll
    for (int o = 16; o > 0; o >>= 1) v += __shfl_xor(v, o, 32);
    return v;
}

__device__ __forceinline__ unsigned int pack_bf16(float a, float b) {
    unsigned short ua = __builtin_bit_cast(unsigned short, (__bf16)a);
    unsigned short ub = __builtin_bit_cast(unsigned short, (__bf16)b);
    return (unsigned int)ua | ((unsigned int)ub << 16);
}

// ---------------------------------------------------------------------------
// CSR build step 1: in-degree counts (int atomics, 1.6M total).
// ---------------------------------------------------------------------------
__global__ void count_kernel(const int* __restrict__ dst, int* __restrict__ cnt) {
    int e = blockIdx.x * blockDim.x + threadIdx.x;
    if (e < N_EDGES)
        __hip_atomic_fetch_add(&cnt[dst[e]], 1, __ATOMIC_RELAXED, __HIP_MEMORY_SCOPE_AGENT);
}

// ---------------------------------------------------------------------------
// CSR build step 2: exclusive scan of counts -> rowptr (3-stage).
// ---------------------------------------------------------------------------
__global__ void scan1_kernel(const int* __restrict__ cnt, int* __restrict__ rowptr,
                             int* __restrict__ bsum) {
    __shared__ int s[SCAN_B];
    int t = threadIdx.x;
    int i = blockIdx.x * SCAN_B + t;
    int v = (i < N_NODES) ? cnt[i] : 0;
    s[t] = v;
    __syncthreads();
#pragma unroll
    for (int o = 1; o < SCAN_B; o <<= 1) {
        int u = (t >= o) ? s[t - o] : 0;
        __syncthreads();
        s[t] += u;
        __syncthreads();
    }
    if (i < N_NODES) rowptr[i] = s[t] - v;         // exclusive within block
    if (t == SCAN_B - 1) bsum[blockIdx.x] = s[t];  // block total
}

__global__ void scan2_kernel(int* __restrict__ bsum) {
    __shared__ int s[512];
    int t = threadIdx.x;
    int v = (t < NBLK) ? bsum[t] : 0;
    s[t] = v;
    __syncthreads();
#pragma unroll
    for (int o = 1; o < 512; o <<= 1) {
        int u = (t >= o) ? s[t - o] : 0;
        __syncthreads();
        s[t] += u;
        __syncthreads();
    }
    if (t < NBLK) bsum[t] = s[t] - v;              // exclusive block offsets
}

__global__ void scan3_kernel(int* __restrict__ rowptr, const int* __restrict__ bsum) {
    int i = blockIdx.x * SCAN_B + threadIdx.x;
    if (i < N_NODES) rowptr[i] += bsum[blockIdx.x];
    if (i == 0) rowptr[N_NODES] = N_EDGES;
}

// ---------------------------------------------------------------------------
// CSR build step 3: bucket fill (src lists per dst node).
// ---------------------------------------------------------------------------
__global__ void fill_kernel(const int* __restrict__ src, const int* __restrict__ dst,
                            const int* __restrict__ rowptr, int* __restrict__ cursor,
                            int* __restrict__ eidx) {
    int e = blockIdx.x * blockDim.x + threadIdx.x;
    if (e < N_EDGES) {
        int d = dst[e];
        int p = __hip_atomic_fetch_add(&cursor[d], 1, __ATOMIC_RELAXED,
                                       __HIP_MEMORY_SCOPE_AGENT);
        eidx[rowptr[d] + p] = src[e];
    }
}

// ---------------------------------------------------------------------------
// LayerNorm: wave32 per row (8 rows / 256-thread block), float4 per lane.
// Safe in-place.
// ---------------------------------------------------------------------------
__global__ void ln_kernel(const float* __restrict__ x, float* __restrict__ h,
                          const float* __restrict__ w, const float* __restrict__ b) {
    int wid  = threadIdx.x >> 5;
    int lane = threadIdx.x & 31;
    int row  = blockIdx.x * 8 + wid;
    const float4 v = ((const float4*)(x + (size_t)row * DIM))[lane];
    float mean = warp_sum32(v.x + v.y + v.z + v.w) * (1.0f / DIM);
    float dx = v.x - mean, dy = v.y - mean, dz = v.z - mean, dw = v.w - mean;
    float var = warp_sum32(dx*dx + dy*dy + dz*dz + dw*dw) * (1.0f / DIM);
    float r = rsqrtf(var + 1e-5f);
    float4 wv = ((const float4*)w)[lane];
    float4 bv = ((const float4*)b)[lane];
    float4 o;
    o.x = dx * r * wv.x + bv.x;
    o.y = dy * r * wv.y + bv.y;
    o.z = dz * r * wv.z + bv.z;
    o.w = dw * r * wv.w + bv.w;
    ((float4*)(h + (size_t)row * DIM))[lane] = o;
}

// ---------------------------------------------------------------------------
// Weight prep: f32 [l][k][n] -> bf16 swizzled into the 16-bit B-fragment VGPR
// layout of v_wmma_f32_16x16x32_bf16.
// Index: ((((l*2+mat)*8 + tile)*4 + kchunk)*32 + lane)*8 + v
// ---------------------------------------------------------------------------
__global__ void prep_w_kernel(const float* __restrict__ Wl, const float* __restrict__ Wr,
                              unsigned int* __restrict__ wswz) {
    int i    = blockIdx.x * blockDim.x + threadIdx.x;   // 0..32767
    int v    = i & 7;
    int lane = (i >> 3) & 31;
    int kc   = (i >> 8) & 3;
    int t    = (i >> 10) & 7;
    int mat  = (i >> 13) & 1;
    int l    = (i >> 14) & 1;
    int n    = t * 16 + (lane & 15);
    int kb   = kc * 32 + ((lane >> 4) ? 8 : 0);
    int k0   = (v < 4) ? (kb + 2 * v) : (kb + 16 + 2 * (v - 4));
    const float* W = mat ? Wr : Wl;
    float a = W[((size_t)l * DIM + k0) * DIM + n];
    float b = W[((size_t)l * DIM + k0 + 1) * DIM + n];
    wswz[i] = pack_bf16(a, b);
}

// ---------------------------------------------------------------------------
// Build a 16x32 bf16 A fragment from an LDS tile (bf16, ROWSTR dwords/row).
// Lane: m = lane&15, kb = kc*32 + (lane>>4)*8; elems 0..7 = K kb..kb+7,
// elems 8..15 = K kb+16..kb+23.  Two ds_load_b128 per fragment.
// ---------------------------------------------------------------------------
__device__ __forceinline__ v16bf lds_frag(const unsigned int* __restrict__ buf,
                                          int m, int kb) {
    const uint4* p = (const uint4*)(buf + m * ROWSTR + (kb >> 1));
    uint4 lo = p[0];
    uint4 hi = p[2];   // +8 dwords = K offset +16
    v8u u;
    u[0] = lo.x; u[1] = lo.y; u[2] = lo.z; u[3] = lo.w;
    u[4] = hi.x; u[5] = hi.y; u[6] = hi.z; u[7] = hi.w;
    return __builtin_bit_cast(v16bf, u);
}

// ---------------------------------------------------------------------------
// Fused SAGE layer: per block = 16 output rows x 128 cols (8 waves).
// Phase 1: each wave gathers+means the in-neighborhoods of 2 nodes (CSR walk,
//          L2-resident row gathers), deg-scales, converts to bf16 in LDS; also
//          stages the 16 h rows as bf16 in LDS.
// Phase 2: wave t computes tile cols [16t,16t+16): 4 k-chunks x 2 GEMMs =
//          8 x v_wmma_f32_16x16x32_bf16, A from LDS (ds_load_b128), B from the
//          swizzled weight table; epilogue: +bias, ReLU, +residual.
// Control flow is wave-uniform -> EXEC all-ones at every WMMA.
// ---------------------------------------------------------------------------
__global__ void sage_layer_kernel(const float* __restrict__ h,
                                  const int* __restrict__ rowptr,
                                  const int* __restrict__ eidx,
                                  const unsigned int* __restrict__ wswz,
                                  const float* __restrict__ bias,
                                  const float* __restrict__ xres,
                                  float* __restrict__ xout, int layer) {
    __shared__ unsigned int aggB[16 * ROWSTR];
    __shared__ unsigned int hB[16 * ROWSTR];
    int wid  = threadIdx.x >> 5;
    int lane = threadIdx.x & 31;
    int rb   = blockIdx.x;

    // ---- Phase 1: stage h + gather-aggregate into LDS (bf16) ----
#pragma unroll
    for (int rr = 0; rr < 2; ++rr) {
        int row  = wid * 2 + rr;          // 0..15 (uniform per wave)
        int node = rb * 16 + row;
        float4 hv = ((const float4*)(h + (size_t)node * DIM))[lane];
        hB[row * ROWSTR + lane * 2]     = pack_bf16(hv.x, hv.y);
        hB[row * ROWSTR + lane * 2 + 1] = pack_bf16(hv.z, hv.w);

        int e0 = rowptr[node], e1 = rowptr[node + 1];
        float4 acc = make_float4(0.f, 0.f, 0.f, 0.f);
        for (int e = e0; e < e1; ++e) {
            int s = eidx[e];
            float4 mv = ((const float4*)(h + (size_t)s * DIM))[lane];
            acc.x += mv.x; acc.y += mv.y; acc.z += mv.z; acc.w += mv.w;
        }
        float rs = 1.0f / fmaxf((float)(e1 - e0), 1.0f);
        aggB[row * ROWSTR + lane * 2]     = pack_bf16(acc.x * rs, acc.y * rs);
        aggB[row * ROWSTR + lane * 2 + 1] = pack_bf16(acc.z * rs, acc.w * rs);
    }
    __syncthreads();

    // ---- Phase 2: WMMA ----
    int t     = wid;                       // n-tile 0..7
    int m     = lane & 15;
    int khalf = (lane >> 4) * 8;
    const unsigned int* wl = wswz + (size_t)((((layer * 2 + 0) * 8 + t) * 4) * 32) * 8;
    const unsigned int* wr = wswz + (size_t)((((layer * 2 + 1) * 8 + t) * 4) * 32) * 8;

    v8f c = {};
#pragma unroll
    for (int kc = 0; kc < 4; ++kc) {
        int kb = kc * 32 + khalf;
        v16bf af = lds_frag(aggB, m, kb);
        v16bf bl16 = __builtin_bit_cast(v16bf, *(const v8u*)(wl + ((size_t)kc * 32 + lane) * 8));
        c = __builtin_amdgcn_wmma_f32_16x16x32_bf16(false, af, false, bl16,
                                                    (short)0, c, false, false);
        v16bf hf = lds_frag(hB, m, kb);
        v16bf br16 = __builtin_bit_cast(v16bf, *(const v8u*)(wr + ((size_t)kc * 32 + lane) * 8));
        c = __builtin_amdgcn_wmma_f32_16x16x32_bf16(false, hf, false, br16,
                                                    (short)0, c, false, false);
    }

    // C/D layout: vgpr j -> M = j + (lane>>4)*8, N = lane&15.
    int col   = t * 16 + m;
    int rbase = rb * 16 + (lane >> 4) * 8;
    float bcol = bias[col];
#pragma unroll
    for (int j = 0; j < 8; ++j) {
        size_t idx = (size_t)(rbase + j) * DIM + col;
        xout[idx] = fmaxf(c[j] + bcol, 0.0f) + xres[idx];
    }
}

__global__ void mask_kernel(float* __restrict__ mask) {
    int i = blockIdx.x * blockDim.x + threadIdx.x;
    if (i < N_NODES) mask[i] = 1.0f;
}

// ---------------------------------------------------------------------------
extern "C" void kernel_launch(void* const* d_in, const int* in_sizes, int n_in,
                              void* d_out, int out_size, void* d_ws, size_t ws_size,
                              hipStream_t stream) {
    const float* node_feature = (const float*)d_in[0];
    const int*   edge_index   = (const int*)d_in[1];
    const int*   src          = edge_index;             // [E]
    const int*   dst          = edge_index + N_EDGES;   // [E]
    const float* ln_w         = (const float*)d_in[4];
    const float* ln_b         = (const float*)d_in[5];
    const float* Wl           = (const float*)d_in[6];
    const float* bl           = (const float*)d_in[7];
    const float* Wr           = (const float*)d_in[8];
    const float* final_ln_w   = (const float*)d_in[9];
    const float* final_ln_b   = (const float*)d_in[10];

    float* dense = (float*)d_out;                       // [N, D] f32 (also x buffer)
    float* mask  = dense + (size_t)N_NODES * DIM;       // [N] (all-true -> 1.0f)

    // Workspace (~59 MB): h, CSR (rowptr/eidx/cnt/cursor/bsum), swizzled weights.
    char* ws = (char*)d_ws;
    float* h        = (float*)ws;        ws += (size_t)N_NODES * DIM * sizeof(float);
    int*   rowptr   = (int*)ws;          ws += (size_t)(N_NODES + 1) * sizeof(int);
    int*   eidx     = (int*)ws;          ws += (size_t)N_EDGES * sizeof(int);
    int*   cnt      = (int*)ws;          ws += (size_t)N_NODES * sizeof(int);
    int*   cursor   = (int*)ws;          ws += (size_t)N_NODES * sizeof(int);
    int*   bsum     = (int*)ws;          ws += (size_t)NBLK * sizeof(int);
    unsigned int* wswz = (unsigned int*)ws;  // 32768 dwords

    // CSR build (graph static across layers)
    hipMemsetAsync(cnt, 0, (size_t)N_NODES * sizeof(int), stream);
    hipMemsetAsync(cursor, 0, (size_t)N_NODES * sizeof(int), stream);
    count_kernel<<<(N_EDGES + 255) / 256, 256, 0, stream>>>(dst, cnt);
    scan1_kernel<<<NBLK, SCAN_B, 0, stream>>>(cnt, rowptr, bsum);
    scan2_kernel<<<1, 512, 0, stream>>>(bsum);
    scan3_kernel<<<NBLK, SCAN_B, 0, stream>>>(rowptr, bsum);
    fill_kernel<<<(N_EDGES + 255) / 256, 256, 0, stream>>>(src, dst, rowptr, cursor, eidx);
    prep_w_kernel<<<32768 / 256, 256, 0, stream>>>(Wl, Wr, wswz);

    const float* x = node_feature;
    for (int l = 0; l < NLAYERS; ++l) {
        ln_kernel<<<N_NODES / 8, 256, 0, stream>>>(x, h, ln_w + l * DIM, ln_b + l * DIM);
        sage_layer_kernel<<<N_NODES / 16, 256, 0, stream>>>(h, rowptr, eidx, wswz,
                                                            bl + l * DIM, x, dense, l);
        x = dense;
    }
    ln_kernel<<<N_NODES / 8, 256, 0, stream>>>(dense, dense, final_ln_w, final_ln_b);
    mask_kernel<<<(N_NODES + 255) / 256, 256, 0, stream>>>(mask);

    (void)in_sizes; (void)n_in; (void)out_size; (void)ws_size;
}
